// KANGuard_45921790329237
// MI455X (gfx1250) — compile-verified, matrix operand order checked by reading, not run on hardware
//
#include <hip/hip_runtime.h>
#include <cstdint>
#include <cstddef>

// ---------------------------------------------------------------------------
// KAN-GNN forward for MI455X (gfx1250, wave32, WMMA).
//   L1: h1 = relu([mean-agg(x) | x] @ [W1l;W1r]^T + b1)     K=256, OUT=128
//   L2: h2 = relu([mean-agg(h1)| h1] @ [W2l;W2r]^T + b2)    K=256, OUT=128
//   K1: h3 = [silu(h2) | bspl(h2)] @ [kb1;ks1]^T            K=1152, OUT=64
//   K2: out = scalar KAN (OUT=2 -> VALU, not worth a WMMA tile)
// Dense math uses v_wmma_f32_16x16x32_f16 (f16 in, f32 accumulate).
// A m-tile is staged in LDS (shared by all n-tile waves of the block);
// B fragments stream from global (each W element read once per block anyway).
// ---------------------------------------------------------------------------

typedef __attribute__((ext_vector_type(16))) _Float16 v16h;
typedef __attribute__((ext_vector_type(8)))  _Float16 v8h;
typedef __attribute__((ext_vector_type(8)))  float    v8f;

// Cox-de Boor, order-3, uniform extended grid on [-1,1]: knots g[i]=(i-3)*0.4-1
static __device__ __forceinline__ void bsplines8(float x, float b3[8]) {
    float g[12];
#pragma unroll
    for (int i = 0; i < 12; ++i) g[i] = (float)(i - 3) * 0.4f - 1.0f;
    float b0[11];
#pragma unroll
    for (int j = 0; j < 11; ++j) b0[j] = (x >= g[j] && x < g[j + 1]) ? 1.0f : 0.0f;
    float b1[10];
#pragma unroll
    for (int j = 0; j < 10; ++j)
        b1[j] = (x - g[j]) * 2.5f      * b0[j] + (g[j + 2] - x) * 2.5f      * b0[j + 1];
    float b2[9];
#pragma unroll
    for (int j = 0; j < 9; ++j)
        b2[j] = (x - g[j]) * 1.25f     * b1[j] + (g[j + 3] - x) * 1.25f     * b1[j + 1];
#pragma unroll
    for (int j = 0; j < 8; ++j)
        b3[j] = (x - g[j]) * (1.f/1.2f)* b2[j] + (g[j + 4] - x) * (1.f/1.2f)* b2[j + 1];
}

static __device__ __forceinline__ float silu(float v) {
    return v / (1.0f + __expf(-v));
}

// ------------------------------ utility kernels ----------------------------

__global__ void k_zero(float* __restrict__ p, int n) {
    int t = blockIdx.x * blockDim.x + threadIdx.x;
    if (t < n) p[t] = 0.0f;
}

__global__ void k_count(const long long* __restrict__ ei, float* __restrict__ cnt,
                        int n_edges) {
    int e = blockIdx.x * blockDim.x + threadIdx.x;
    if (e < n_edges) atomicAdd(&cnt[(int)ei[(size_t)n_edges + e]], 1.0f);
}

// one thread = (edge, 4 channels); gathers hit L2 (node table = 25.6MB << 192MB L2)
__global__ void k_scatter(const float* __restrict__ feat, const long long* __restrict__ ei,
                          float* __restrict__ agg, int n_edges) {
    int t = blockIdx.x * blockDim.x + threadIdx.x;
    int e = t >> 5;
    if (e >= n_edges) return;
    int c4 = t & 31;
    int src = (int)ei[e];
    int dst = (int)ei[(size_t)n_edges + e];
    float4 v = ((const float4*)feat)[(size_t)src * 32 + c4];
    float* d = agg + (size_t)dst * 128 + c4 * 4;
    atomicAdd(d + 0, v.x); atomicAdd(d + 1, v.y);
    atomicAdd(d + 2, v.z); atomicAdd(d + 3, v.w);
}

// A16 row (K=256) = [ agg/deg (f16) | feat (f16) ]
__global__ void k_build_a(const float* __restrict__ agg, const float* __restrict__ cnt,
                          const float* __restrict__ feat, _Float16* __restrict__ A16,
                          int n_nodes) {
    int t = blockIdx.x * blockDim.x + threadIdx.x;
    if (t >= n_nodes * 128) return;
    int node = t >> 7, c = t & 127;
    float d = fmaxf(cnt[node], 1.0f);
    _Float16* row = A16 + (size_t)node * 256;
    row[c]       = (_Float16)(agg[t] / d);
    row[128 + c] = (_Float16)feat[t];
}

// W16 row o (K=256) = [ wl[o,:] | wr[o,:] ]  (out-major rows == B columns)
__global__ void k_prep_w_pair(const float* __restrict__ wl, const float* __restrict__ wr,
                              _Float16* __restrict__ W16, int out_ch) {
    int t = blockIdx.x * blockDim.x + threadIdx.x;
    if (t >= out_ch * 256) return;
    int o = t >> 8, j = t & 255;
    W16[t] = (_Float16)(j < 128 ? wl[o * 128 + j] : wr[o * 128 + (j - 128)]);
}

// W16 row o (K=in*9) = [ base_w[o,:] | spline_w[o,:,:] flattened ]
__global__ void k_prep_w_kan(const float* __restrict__ basew, const float* __restrict__ splinew,
                             _Float16* __restrict__ W16, int out_ch, int in_ch) {
    int rowlen = in_ch * 9;
    int t = blockIdx.x * blockDim.x + threadIdx.x;
    if (t >= out_ch * rowlen) return;
    int o = t / rowlen, j = t % rowlen;
    W16[t] = (_Float16)(j < in_ch ? basew[(size_t)o * in_ch + j]
                                  : splinew[(size_t)o * in_ch * 8 + (j - in_ch)]);
}

// KAN feature rows (K=1152): [ silu(h) (128) | bases (128*8) ]
__global__ void k_kan_feat(const float* __restrict__ h, _Float16* __restrict__ F,
                           int n_nodes) {
    int t = blockIdx.x * blockDim.x + threadIdx.x;
    if (t >= n_nodes * 128) return;
    int node = t >> 7, c = t & 127;
    float v = h[t];
    _Float16* row = F + (size_t)node * 1152;
    row[c] = (_Float16)silu(v);
    float b[8];
    bsplines8(v, b);
    v8h hb;
#pragma unroll
    for (int k = 0; k < 8; ++k) hb[k] = (_Float16)b[k];
    *(v8h*)(row + 128 + c * 8) = hb;   // 16B-aligned packed store
}

// ------------------------------ WMMA GEMM ----------------------------------
// out = A(MxK,f16) * W(OUTxK,f16)^T. Block = OUT/16 waves, one 16-row m-tile.
// The 16xK A tile (contiguous in memory, rows have stride K) is staged in LDS
// once and shared by all waves (8x/4x L2-traffic cut on the A side).
// Fragment layout (contiguous k, not ISA table order): lane l&15 = row(A)/column(B), l>>4 selects K-half;
// each lane reads 16 contiguous f16 -> pure ds_load / global_load, no shuffles.
template <int K, int OUT, bool RELU, bool HASBIAS>
__global__ __launch_bounds__(256) void k_wmma_gemm(const _Float16* __restrict__ A,
                                                   const _Float16* __restrict__ W,
                                                   const float* __restrict__ bias,
                                                   float* __restrict__ out, int M) {
    constexpr int NT = OUT / 16;             // waves per block
    __shared__ _Float16 As[16 * K];          // 8KB (K=256) / 36KB (K=1152)

    const int lane = threadIdx.x;            // 0..31
    const int wid  = threadIdx.y;            // n-tile index
    const int m0   = blockIdx.x * 16;

    // cooperative linear copy of the 16xK tile into LDS (uint4 = 8 halves)
    {
        const uint4* src = (const uint4*)(A + (size_t)m0 * K);
        uint4*       dst = (uint4*)As;
        int vrows  = M - m0; if (vrows > 16) vrows = 16;
        int vchunk = vrows * (K / 8);
        int tid    = wid * 32 + lane;
        for (int i = tid; i < 16 * (K / 8); i += 32 * NT)
            if (i < vchunk) dst[i] = src[i];
    }
    __syncthreads();

    const int r  = lane & 15;
    const int kh = (lane >> 4) * 16;
    const _Float16* ap = As + r * K + kh;
    const _Float16* bp = W + (size_t)(wid * 16 + r) * K + kh;

    v8f acc = {};
#pragma unroll 4
    for (int k = 0; k < K; k += 32) {
        v16h a = *(const v16h*)(ap + k);     // ds_load, 32B aligned
        v16h b = *(const v16h*)(bp + k);     // global_load_b128 x2
        acc = __builtin_amdgcn_wmma_f32_16x16x32_f16(false, a, false, b,
                                                     (short)0, acc, false, false);
    }

    // epilogue: D VGPR i holds row m0 + i (+8 for hi lanes), column = lane&15
    const int n     = lane & 15;
    const int mbase = m0 + ((lane >> 4) << 3);
    float bv = HASBIAS ? bias[wid * 16 + n] : 0.0f;
    float* op = out + (size_t)mbase * OUT + wid * 16 + n;
    if (m0 + 16 <= M) {                      // full tile: no per-element guards
#pragma unroll
        for (int i = 0; i < 8; ++i) {
            float v = acc[i] + bv;
            if (RELU) v = fmaxf(v, 0.0f);
            op[(size_t)i * OUT] = v;
        }
    } else {                                 // tail tile (not hit for M=50000)
#pragma unroll
        for (int i = 0; i < 8; ++i) {
            if (mbase + i < M) {
                float v = acc[i] + bv;
                if (RELU) v = fmaxf(v, 0.0f);
                op[(size_t)i * OUT] = v;
            }
        }
    }
}

// KAN2: OUT=2, K=576 -> scalar per node (WMMA tile would be 7/8 wasted)
__global__ void k_kan2(const float* __restrict__ h3, const float* __restrict__ basew,
                       const float* __restrict__ splinew, float* __restrict__ out,
                       int n_nodes) {
    int node = blockIdx.x * blockDim.x + threadIdx.x;
    if (node >= n_nodes) return;
    const float* hr = h3 + (size_t)node * 64;
    float a0 = 0.0f, a1 = 0.0f;
    for (int c = 0; c < 64; ++c) {
        float v = hr[c];
        float s = silu(v);
        a0 += s * basew[c];
        a1 += s * basew[64 + c];
        float b[8];
        bsplines8(v, b);
#pragma unroll
        for (int k = 0; k < 8; ++k) {
            a0 += b[k] * splinew[c * 8 + k];
            a1 += b[k] * splinew[512 + c * 8 + k];
        }
    }
    out[node * 2 + 0] = a0;
    out[node * 2 + 1] = a1;
}

// ------------------------------ launcher -----------------------------------

extern "C" void kernel_launch(void* const* d_in, const int* in_sizes, int n_in,
                              void* d_out, int out_size, void* d_ws, size_t ws_size,
                              hipStream_t stream) {
    (void)n_in; (void)out_size; (void)ws_size;
    const float*     x   = (const float*)d_in[0];
    const long long* ei  = (const long long*)d_in[1];
    const float*     W1l = (const float*)d_in[2];
    const float*     b1  = (const float*)d_in[3];
    const float*     W1r = (const float*)d_in[4];
    const float*     W2l = (const float*)d_in[5];
    const float*     b2  = (const float*)d_in[6];
    const float*     W2r = (const float*)d_in[7];
    const float*     kb1 = (const float*)d_in[8];
    const float*     ks1 = (const float*)d_in[9];
    const float*     kb2 = (const float*)d_in[10];
    const float*     ks2 = (const float*)d_in[11];
    float*           out = (float*)d_out;

    const int n_nodes = in_sizes[0] / 128;
    const int n_edges = in_sizes[1] / 2;

    // scratch carve-up (256B aligned regions)
    char* w = (char*)d_ws;
    auto take = [&](size_t bytes) {
        char* p = w;
        w += (bytes + 255) & ~(size_t)255;
        return p;
    };
    float*    cnt = (float*)take((size_t)n_nodes * 4);
    float*    agg = (float*)take((size_t)n_nodes * 128 * 4);
    float*    h1  = (float*)take((size_t)n_nodes * 128 * 4);
    float*    h2  = (float*)take((size_t)n_nodes * 128 * 4);
    float*    h3  = (float*)take((size_t)n_nodes * 64 * 4);
    _Float16* A16 = (_Float16*)take((size_t)n_nodes * 1152 * 2);  // reused K=256/1152
    _Float16* W16 = (_Float16*)take((size_t)64 * 1152 * 2);       // reused per phase

    auto cdiv = [](int a, int b) { return (a + b - 1) / b; };
    const int TB = 256;
    const int mtiles = cdiv(n_nodes, 16);

    // ---- layer 1 ----
    k_zero<<<cdiv(n_nodes, TB), TB, 0, stream>>>(cnt, n_nodes);
    k_zero<<<cdiv(n_nodes * 128, TB), TB, 0, stream>>>(agg, n_nodes * 128);
    k_count<<<cdiv(n_edges, TB), TB, 0, stream>>>(ei, cnt, n_edges);
    k_scatter<<<cdiv(n_edges * 32, TB), TB, 0, stream>>>(x, ei, agg, n_edges);
    k_prep_w_pair<<<cdiv(128 * 256, TB), TB, 0, stream>>>(W1l, W1r, W16, 128);
    k_build_a<<<cdiv(n_nodes * 128, TB), TB, 0, stream>>>(agg, cnt, x, A16, n_nodes);
    {
        dim3 blk(32, 8);  // 8 waves = 8 column tiles of OUT=128
        k_wmma_gemm<256, 128, true, true><<<mtiles, blk, 0, stream>>>(A16, W16, b1, h1, n_nodes);
    }

    // ---- layer 2 ----
    k_zero<<<cdiv(n_nodes * 128, TB), TB, 0, stream>>>(agg, n_nodes * 128);
    k_scatter<<<cdiv(n_edges * 32, TB), TB, 0, stream>>>(h1, ei, agg, n_edges);
    k_prep_w_pair<<<cdiv(128 * 256, TB), TB, 0, stream>>>(W2l, W2r, W16, 128);
    k_build_a<<<cdiv(n_nodes * 128, TB), TB, 0, stream>>>(agg, cnt, h1, A16, n_nodes);
    {
        dim3 blk(32, 8);
        k_wmma_gemm<256, 128, true, true><<<mtiles, blk, 0, stream>>>(A16, W16, b2, h2, n_nodes);
    }

    // ---- KAN layer 1 (HID=128 -> 64), spline term as K=1152 GEMM ----
    k_prep_w_kan<<<cdiv(64 * 1152, TB), TB, 0, stream>>>(kb1, ks1, W16, 64, 128);
    k_kan_feat<<<cdiv(n_nodes * 128, TB), TB, 0, stream>>>(h2, A16, n_nodes);
    {
        dim3 blk(32, 4);  // 4 waves = 4 column tiles of OUT=64
        k_wmma_gemm<1152, 64, false, false><<<mtiles, blk, 0, stream>>>(A16, W16, nullptr, h3, n_nodes);
    }

    // ---- KAN layer 2 (64 -> 2), scalar ----
    k_kan2<<<cdiv(n_nodes, TB), TB, 0, stream>>>(h3, kb2, ks2, out, n_nodes);
}
